// AttentionCircuit_88124138979397
// MI455X (gfx1250) — matrix operand order repaired and checked by reading, not compiled, not run on hardware
//
#include <hip/hip_runtime.h>
#include <stdint.h>
#include <stddef.h>

// ---------------------------------------------------------------------------
// Types for CDNA5 WMMA (wave32): D(f32 16x16) = A(bf16 16x32) * B(bf16 32x16) + C
// ---------------------------------------------------------------------------
typedef __bf16 bf16_t;
typedef __attribute__((ext_vector_type(16))) __bf16 v16bf;
typedef __attribute__((ext_vector_type(8)))  __bf16 v8bf;
typedef __attribute__((ext_vector_type(8)))  float  v8f;

#define TOK    4096   // B*S
#define DMODEL 1024
#define RDIM   128
#define NPOOL  8
#define NHEADS 16
#define DHEAD  64
#define SEQ    2048

__device__ __forceinline__ v8f wmma_bf(v16bf a, v16bf b, v8f c) {
  // (neg_a, A, neg_b, B, c_mod, C, reuse_a, reuse_b)
  return __builtin_amdgcn_wmma_f32_16x16x32_bf16(false, a, false, b, (short)0, c,
                                                 false, false);
}

__device__ __forceinline__ v8f vzero8() {
  v8f z;
#pragma unroll
  for (int i = 0; i < 8; i++) z[i] = 0.0f;
  return z;
}

// A fragment (16x32, bf16), row-major source with leading dim `ld`.
// Layout used here (contiguous k): lane L holds row M=L&15; element e -> K = e + (L>=16?16:0)
// => two contiguous 8-element chunks at k0+(L>=16?16:0) and +8.
__device__ __forceinline__ v16bf load_afrag(const bf16_t* __restrict__ base, int ld,
                                            int rowBase, int k0) {
  int lane = threadIdx.x & 31;
  int m  = rowBase + (lane & 15);
  int ko = (lane >> 4) ? 16 : 0;
  const bf16_t* p = base + (size_t)m * ld + k0 + ko;
  v8bf lo = *(const v8bf*)p;
  v8bf hi = *(const v8bf*)(p + 8);
  v16bf a;
#pragma unroll
  for (int e = 0; e < 8; e++) { a[e] = lo[e]; a[e + 8] = hi[e]; }
  return a;
}

// B fragment (32x16, bf16) from a column-major matrix Bt[col][k] with leading dim `ld`.
// Layout used here (contiguous k): lane L holds col N=L&15; element e -> K = e + (L>=16?16:0)
// => one contiguous 16-element (32B) chunk per lane.
__device__ __forceinline__ v16bf load_bfrag(const bf16_t* __restrict__ base, int ld,
                                            int colBase, int k0) {
  int lane = threadIdx.x & 31;
  int col = colBase + (lane & 15);
  int ko  = (lane >> 4) ? 16 : 0;
  return *(const v16bf*)(base + (size_t)col * ld + k0 + ko);
}

// ---------------------------------------------------------------------------
// Pack / convert kernels (fp32 -> bf16, with layout permutes)
// ---------------------------------------------------------------------------
__global__ void __launch_bounds__(256) k_cvt(const float* __restrict__ in,
                                             bf16_t* __restrict__ out, int n) {
  int i = blockIdx.x * blockDim.x + threadIdx.x;
  if (i < n) out[i] = (bf16_t)in[i];
}

// feature neurons [N,D,R] -> Bt[(n*R+r)][d]   (column-major B for X @ neurons)
__global__ void __launch_bounds__(256) k_pack_feat(const float* __restrict__ in,
                                                   bf16_t* __restrict__ out) {
  int idx = blockIdx.x * blockDim.x + threadIdx.x;      // < N*R*D = 1M
  int col = idx >> 10, d = idx & 1023;
  int n = col >> 7,   r = col & 127;
  out[idx] = (bf16_t)in[((size_t)n * DMODEL + d) * RDIM + r];
}

// restore neurons [N,R,D] -> Rt[(n*D+d)][r]   (column-major B for h @ neurons)
__global__ void __launch_bounds__(256) k_pack_rest(const float* __restrict__ in,
                                                   bf16_t* __restrict__ out) {
  int idx = blockIdx.x * blockDim.x + threadIdx.x;      // < N*D*R = 1M
  int col = idx >> 7, r = idx & 127;
  int n = col >> 10,  d = col & 1023;
  out[idx] = (bf16_t)in[((size_t)n * RDIM + r) * DMODEL + d];
}

// W_O [Din,Dout] -> WOt[dout][din]
__global__ void __launch_bounds__(256) k_pack_wo(const float* __restrict__ in,
                                                 bf16_t* __restrict__ out) {
  int idx = blockIdx.x * blockDim.x + threadIdx.x;      // < 1M
  int j = idx >> 10, din = idx & 1023;
  out[idx] = (bf16_t)in[(size_t)din * DMODEL + j];
}

// mixture weights [T,N] -> [N,T]  (so epilogues do one 32B vector load per n)
__global__ void __launch_bounds__(256) k_pack_wt(const float* __restrict__ in,
                                                 float* __restrict__ out) {
  int idx = blockIdx.x * blockDim.x + threadIdx.x;      // < N*T = 32768
  int n = idx >> 12, t = idx & 4095;
  out[idx] = in[(size_t)t * NPOOL + n];
}

// ---------------------------------------------------------------------------
// Feature GEMM with fused mixture epilogue:
//   h[t,r] = sum_n w[n,t] * (X @ A_n)[t,r]
// k0-outer / n-inner: one A fragment feeds 8 WMMAs (8 accumulators).
// One wave per 16x16 tile of h [4096 x 128]; 2048 tiles, 4 waves/block.
// ---------------------------------------------------------------------------
template <int COMPUTE_K>
__global__ void __launch_bounds__(128)
k_feature(const bf16_t* __restrict__ XB, const bf16_t* __restrict__ Bt,
          const float* __restrict__ wQt, const float* __restrict__ wKt,
          bf16_t* __restrict__ HQ, bf16_t* __restrict__ HK) {
  int wave = threadIdx.x >> 5, lane = threadIdx.x & 31;
  int gw = blockIdx.x * 4 + wave;                  // 0..2047
  int tokenBase = (gw >> 3) * 16;
  int rBase     = (gw & 7) * 16;
  int hi = lane >> 4, l15 = lane & 15;

  v8f accs[NPOOL];
#pragma unroll
  for (int n = 0; n < NPOOL; n++) accs[n] = vzero8();

  for (int k0 = 0; k0 < DMODEL; k0 += 32) {
    v16bf a = load_afrag(XB, DMODEL, tokenBase, k0);
#pragma unroll
    for (int n = 0; n < NPOOL; n++) {
      v16bf b = load_bfrag(Bt, DMODEL, n * RDIM + rBase, k0);
      accs[n] = wmma_bf(a, b, accs[n]);
    }
  }

  v8f fq = vzero8(), fk = vzero8();
  int trow = tokenBase + hi * 8;                   // C layout: row = i + 8*(lane>=16)
#pragma unroll
  for (int n = 0; n < NPOOL; n++) {
    v8f wv = *(const v8f*)(wQt + (size_t)n * TOK + trow);
#pragma unroll
    for (int i = 0; i < 8; i++) fq[i] += wv[i] * accs[n][i];
    if (COMPUTE_K) {
      v8f wk = *(const v8f*)(wKt + (size_t)n * TOK + trow);
#pragma unroll
      for (int i = 0; i < 8; i++) fk[i] += wk[i] * accs[n][i];
    }
  }

  int col = rBase + l15;
#pragma unroll
  for (int i = 0; i < 8; i++) {
    HQ[(size_t)(trow + i) * RDIM + col] = (bf16_t)fq[i];
    if (COMPUTE_K) HK[(size_t)(trow + i) * RDIM + col] = (bf16_t)fk[i];
  }
}

// ---------------------------------------------------------------------------
// Restore GEMM with fused mixture epilogue + head-layout store:
//   out[t,d] = sum_n w[n,t] * (h @ B_n)[t,d]
// VLAYOUT==0: write [b,h,s,dh] (Q/K);  VLAYOUT==1: write d-major [b,h,dh,s] (V).
// 16384 tiles of 16x16 over [4096 x 1024].
// ---------------------------------------------------------------------------
template <int VLAYOUT>
__global__ void __launch_bounds__(128)
k_restore(const bf16_t* __restrict__ H, const bf16_t* __restrict__ Rt,
          const float* __restrict__ wt, bf16_t* __restrict__ outp) {
  int wave = threadIdx.x >> 5, lane = threadIdx.x & 31;
  int gw = blockIdx.x * 4 + wave;                  // 0..16383
  int tokenBase = (gw >> 6) * 16;
  int dBase     = (gw & 63) * 16;
  int hi = lane >> 4, l15 = lane & 15;

  v8f accs[NPOOL];
#pragma unroll
  for (int n = 0; n < NPOOL; n++) accs[n] = vzero8();

#pragma unroll
  for (int k0 = 0; k0 < RDIM; k0 += 32) {
    v16bf a = load_afrag(H, RDIM, tokenBase, k0);
#pragma unroll
    for (int n = 0; n < NPOOL; n++) {
      v16bf b = load_bfrag(Rt, RDIM, n * DMODEL + dBase, k0);
      accs[n] = wmma_bf(a, b, accs[n]);
    }
  }

  v8f fo = vzero8();
  int trow = tokenBase + hi * 8;
#pragma unroll
  for (int n = 0; n < NPOOL; n++) {
    v8f wv = *(const v8f*)(wt + (size_t)n * TOK + trow);
#pragma unroll
    for (int i = 0; i < 8; i++) fo[i] += wv[i] * accs[n][i];
  }

  int d = dBase + l15;
  int h = d >> 6, dq = d & 63;
#pragma unroll
  for (int i = 0; i < 8; i++) {
    int t = trow + i;
    int b_ = t >> 11, s = t & (SEQ - 1);
    if (VLAYOUT == 0)
      outp[(((size_t)(b_ * NHEADS + h) * SEQ + s) << 6) + dq] = (bf16_t)fo[i];
    else
      outp[((size_t)(b_ * NHEADS + h) * DHEAD + dq) * SEQ + s] = (bf16_t)fo[i];
  }
}

// ---------------------------------------------------------------------------
// Causal flash attention. Grid (B*H, S/64), block=128 (4 waves); each wave owns
// 16 query rows, streams keys in blocks of 32. P re-layout C->A goes via LDS.
// ---------------------------------------------------------------------------
__global__ void __launch_bounds__(128)
k_attn(const bf16_t* __restrict__ Q, const bf16_t* __restrict__ K,
       const bf16_t* __restrict__ VT, bf16_t* __restrict__ AO) {
  __shared__ __align__(32) bf16_t Plds[4][16][32];

  int wave = threadIdx.x >> 5, lane = threadIdx.x & 31;
  int hi = lane >> 4, l15 = lane & 15;
  int bh = blockIdx.x;                              // b*16 + h
  int qBase = blockIdx.y * 64 + wave * 16;

  const bf16_t* Qp = Q  + (size_t)bh * SEQ * DHEAD;
  const bf16_t* Kp = K  + (size_t)bh * SEQ * DHEAD;
  const bf16_t* Vp = VT + (size_t)bh * DHEAD * SEQ; // d-major: Vp[d][s]

  v16bf aq0 = load_afrag(Qp, DHEAD, qBase, 0);
  v16bf aq1 = load_afrag(Qp, DHEAD, qBase, 32);

  v8f o0 = vzero8(), o1 = vzero8(), o2 = vzero8(), o3 = vzero8();
  float m8[8], l8[8];
#pragma unroll
  for (int i = 0; i < 8; i++) { m8[i] = -1e30f; l8[i] = 0.0f; }

  for (int kb = 0; kb <= qBase + 15; kb += 32) {
    // S = Q K^T for 32 keys (two 16-col tiles); K row-major == column-major B.
    v8f s0 = vzero8(), s1 = vzero8();
    {
      v16bf b;
      b = load_bfrag(Kp, DHEAD, kb, 0);       s0 = wmma_bf(aq0, b, s0);
      b = load_bfrag(Kp, DHEAD, kb, 32);      s0 = wmma_bf(aq1, b, s0);
      b = load_bfrag(Kp, DHEAD, kb + 16, 0);  s1 = wmma_bf(aq0, b, s1);
      b = load_bfrag(Kp, DHEAD, kb + 16, 32); s1 = wmma_bf(aq1, b, s1);
    }
    int key0 = kb + l15, key1 = key0 + 16;
    float p0[8], p1[8];
#pragma unroll
    for (int i = 0; i < 8; i++) {
      int qrow = qBase + i + hi * 8;
      float v0 = (key0 <= qrow) ? s0[i] * 0.125f : -1e30f;  // 1/sqrt(64)
      float v1 = (key1 <= qrow) ? s1[i] * 0.125f : -1e30f;
      float rmax = fmaxf(v0, v1);
#pragma unroll
      for (int msk = 1; msk <= 8; msk <<= 1)
        rmax = fmaxf(rmax, __shfl_xor(rmax, msk, 32));      // 16-lane half reduce
      float mnew  = fmaxf(m8[i], rmax);
      float alpha = __expf(m8[i] - mnew);
      float e0 = __expf(v0 - mnew);
      float e1 = __expf(v1 - mnew);
      float rsum = e0 + e1;
#pragma unroll
      for (int msk = 1; msk <= 8; msk <<= 1)
        rsum += __shfl_xor(rsum, msk, 32);
      l8[i] = l8[i] * alpha + rsum;
      m8[i] = mnew;
      o0[i] *= alpha; o1[i] *= alpha; o2[i] *= alpha; o3[i] *= alpha;
      p0[i] = e0; p1[i] = e1;
    }
    // Stage P (C layout) into LDS, re-read as a 16x32 A fragment.
#pragma unroll
    for (int i = 0; i < 8; i++) {
      Plds[wave][i + hi * 8][l15]      = (bf16_t)p0[i];
      Plds[wave][i + hi * 8][l15 + 16] = (bf16_t)p1[i];
    }
    asm volatile("s_wait_dscnt 0x0" ::: "memory");
    v16bf ap = load_afrag(&Plds[wave][0][0], 32, 0, 0);
    {
      v16bf b;
      b = load_bfrag(Vp, SEQ, 0,  kb); o0 = wmma_bf(ap, b, o0);
      b = load_bfrag(Vp, SEQ, 16, kb); o1 = wmma_bf(ap, b, o1);
      b = load_bfrag(Vp, SEQ, 32, kb); o2 = wmma_bf(ap, b, o2);
      b = load_bfrag(Vp, SEQ, 48, kb); o3 = wmma_bf(ap, b, o3);
    }
  }

  int b_ = bh >> 4, h = bh & 15;
#pragma unroll
  for (int i = 0; i < 8; i++) {
    float inv = 1.0f / l8[i];
    size_t t = (size_t)b_ * SEQ + qBase + i + hi * 8;
    int dm = h * DHEAD + l15;
    AO[t * DMODEL + dm]      = (bf16_t)(o0[i] * inv);
    AO[t * DMODEL + dm + 16] = (bf16_t)(o1[i] * inv);
    AO[t * DMODEL + dm + 32] = (bf16_t)(o2[i] * inv);
    AO[t * DMODEL + dm + 48] = (bf16_t)(o3[i] * inv);
  }
}

// ---------------------------------------------------------------------------
// Output projection: out[t,j] = sum_d AO[t,d] * W_O[d,j]   (fp32 output)
// ---------------------------------------------------------------------------
__global__ void __launch_bounds__(128)
k_outproj(const bf16_t* __restrict__ AO, const bf16_t* __restrict__ WOt,
          float* __restrict__ out) {
  int wave = threadIdx.x >> 5, lane = threadIdx.x & 31;
  int gw = blockIdx.x * 4 + wave;                  // 0..16383
  int tokenBase = (gw >> 6) * 16;
  int cBase     = (gw & 63) * 16;
  int hi = lane >> 4, l15 = lane & 15;

  v8f acc = vzero8();
#pragma unroll 4
  for (int k0 = 0; k0 < DMODEL; k0 += 32) {
    v16bf a = load_afrag(AO, DMODEL, tokenBase, k0);
    v16bf b = load_bfrag(WOt, DMODEL, cBase, k0);
    acc = wmma_bf(a, b, acc);
  }
  int col = cBase + l15;
#pragma unroll
  for (int i = 0; i < 8; i++)
    out[(size_t)(tokenBase + i + hi * 8) * DMODEL + col] = acc[i];
}

// ---------------------------------------------------------------------------
// Launcher
// ---------------------------------------------------------------------------
extern "C" void kernel_launch(void* const* d_in, const int* in_sizes, int n_in,
                              void* d_out, int out_size, void* d_ws, size_t ws_size,
                              hipStream_t stream) {
  (void)in_sizes; (void)n_in; (void)out_size; (void)ws_size;

  const float* x    = (const float*)d_in[0];
  const float* fqkN = (const float*)d_in[1];
  const float* fvN  = (const float*)d_in[2];
  const float* rqkN = (const float*)d_in[3];
  const float* rvN  = (const float*)d_in[4];
  const float* wQ   = (const float*)d_in[5];
  const float* wK   = (const float*)d_in[6];
  const float* wV   = (const float*)d_in[7];
  const float* rwQ  = (const float*)d_in[8];
  const float* rwK  = (const float*)d_in[9];
  const float* rwV  = (const float*)d_in[10];
  const float* WO   = (const float*)d_in[11];
  float* out = (float*)d_out;

  // Workspace carve (256B aligned slices)
  size_t off = 0;
  char* base = (char*)d_ws;
  auto carve = [&](size_t bytes) -> void* {
    void* p = (void*)(base + off);
    off += (bytes + 255) & ~(size_t)255;
    return p;
  };
  bf16_t* XB   = (bf16_t*)carve((size_t)TOK * DMODEL * 2);          // x, bf16
  bf16_t* FQKt = (bf16_t*)carve((size_t)NPOOL * RDIM * DMODEL * 2); // [n*R+r][d]
  bf16_t* FVt  = (bf16_t*)carve((size_t)NPOOL * RDIM * DMODEL * 2);
  bf16_t* RQKt = (bf16_t*)carve((size_t)NPOOL * DMODEL * RDIM * 2); // [n*D+d][r]
  bf16_t* RVt  = (bf16_t*)carve((size_t)NPOOL * DMODEL * RDIM * 2);
  bf16_t* WOt  = (bf16_t*)carve((size_t)DMODEL * DMODEL * 2);       // [j][d]
  bf16_t* HQ   = (bf16_t*)carve((size_t)TOK * RDIM * 2);
  bf16_t* HK   = (bf16_t*)carve((size_t)TOK * RDIM * 2);
  bf16_t* HV   = (bf16_t*)carve((size_t)TOK * RDIM * 2);
  bf16_t* QA   = (bf16_t*)carve((size_t)2 * NHEADS * SEQ * DHEAD * 2); // [b,h,s,dh]
  bf16_t* KA   = (bf16_t*)carve((size_t)2 * NHEADS * SEQ * DHEAD * 2);
  bf16_t* VT   = (bf16_t*)carve((size_t)2 * NHEADS * DHEAD * SEQ * 2); // [b,h,dh,s]
  bf16_t* AO   = (bf16_t*)carve((size_t)TOK * DMODEL * 2);
  float*  wQt  = (float*)carve((size_t)NPOOL * TOK * 4);  // transposed weights
  float*  wKt  = (float*)carve((size_t)NPOOL * TOK * 4);
  float*  wVt  = (float*)carve((size_t)NPOOL * TOK * 4);
  float*  rwQt = (float*)carve((size_t)NPOOL * TOK * 4);
  float*  rwKt = (float*)carve((size_t)NPOOL * TOK * 4);
  float*  rwVt = (float*)carve((size_t)NPOOL * TOK * 4);

  // 1) pack / convert
  k_cvt      <<<(TOK * DMODEL + 255) / 256, 256, 0, stream>>>(x, XB, TOK * DMODEL);
  k_pack_feat<<<(NPOOL * RDIM * DMODEL + 255) / 256, 256, 0, stream>>>(fqkN, FQKt);
  k_pack_feat<<<(NPOOL * RDIM * DMODEL + 255) / 256, 256, 0, stream>>>(fvN, FVt);
  k_pack_rest<<<(NPOOL * DMODEL * RDIM + 255) / 256, 256, 0, stream>>>(rqkN, RQKt);
  k_pack_rest<<<(NPOOL * DMODEL * RDIM + 255) / 256, 256, 0, stream>>>(rvN, RVt);
  k_pack_wo  <<<(DMODEL * DMODEL + 255) / 256, 256, 0, stream>>>(WO, WOt);
  k_pack_wt  <<<(NPOOL * TOK + 255) / 256, 256, 0, stream>>>(wQ,  wQt);
  k_pack_wt  <<<(NPOOL * TOK + 255) / 256, 256, 0, stream>>>(wK,  wKt);
  k_pack_wt  <<<(NPOOL * TOK + 255) / 256, 256, 0, stream>>>(wV,  wVt);
  k_pack_wt  <<<(NPOOL * TOK + 255) / 256, 256, 0, stream>>>(rwQ, rwQt);
  k_pack_wt  <<<(NPOOL * TOK + 255) / 256, 256, 0, stream>>>(rwK, rwKt);
  k_pack_wt  <<<(NPOOL * TOK + 255) / 256, 256, 0, stream>>>(rwV, rwVt);

  // 2) feature projections (QK pool shared: one GEMM pass, two epilogues)
  k_feature<1><<<512, 128, 0, stream>>>(XB, FQKt, wQt, wKt, HQ, HK);
  k_feature<0><<<512, 128, 0, stream>>>(XB, FVt,  wVt, nullptr, HV, nullptr);

  // 3) restore projections into attention layouts
  k_restore<0><<<4096, 128, 0, stream>>>(HQ, RQKt, rwQt, QA);
  k_restore<0><<<4096, 128, 0, stream>>>(HK, RQKt, rwKt, KA);
  k_restore<1><<<4096, 128, 0, stream>>>(HV, RVt,  rwVt, VT);

  // 4) causal flash attention
  k_attn<<<dim3(2 * NHEADS, SEQ / 64), 128, 0, stream>>>(QA, KA, VT, AO);

  // 5) output projection (fp32 out)
  k_outproj<<<4096, 128, 0, stream>>>(AO, WOt, out);
}
